// GCNBiLSTM_61375082660312
// MI455X (gfx1250) — matrix operand order repaired and checked by reading, not compiled, not run on hardware
//
#include <hip/hip_runtime.h>
#include <hip/hip_bf16.h>
#include <stdint.h>

typedef __bf16 bf16;
typedef __bf16 v16bf __attribute__((ext_vector_type(16)));
typedef float  v8f   __attribute__((ext_vector_type(8)));
typedef unsigned int v4u __attribute__((ext_vector_type(4)));
typedef int    v4i   __attribute__((ext_vector_type(4)));
typedef int    v8i   __attribute__((ext_vector_type(8)));

#define B_   64
#define T_   256
#define NN   61
#define FIN  3
#define GH   64
#define LH   128
#define NCLS 10
#define S_   (B_*T_)        // 16384 samples
#define KSEQ (NN*GH)        // 3904
#define NG2  1024           // 4*LH*2dirs

static __device__ __forceinline__ float sigf(float x){ return 1.0f/(1.0f+expf(-x)); }
static __device__ __forceinline__ float geluf(float v){ return 0.5f*v*(1.0f+erff(v*0.70710678118f)); }

// Fragment-major index for a 64x64 B matrix: per-lane 16 halfs contiguous.
// B-frag layout: col = lane&15 (+16*subtile), k chunk of 32: lanes>=16 hold k+16.
static __device__ __forceinline__ int fidx64(int k, int n){
  int sub = n >> 4, kc = k >> 5;
  int lane = (n & 15) | (((k >> 4) & 1) << 4);
  return ((sub*2 + kc)*32 + lane)*16 + (k & 15);
}

// ---------------------------------------------------------------------------
// TDM: 2D tile load (global -> LDS), bf16 elements (6-arg builtin variant).
// tensor_d0/tensor_d1 give hardware OOB zero-fill for partial tiles.
// ---------------------------------------------------------------------------
static __device__ __forceinline__ void tdm_load_2d(unsigned int lds_addr, const void* gptr,
                                                   unsigned int tile_d0, unsigned int tile_d1,
                                                   unsigned int tensor_d0, unsigned int tensor_d1,
                                                   unsigned int stride_elems)
{
  unsigned long long ga = (unsigned long long)(uintptr_t)gptr;
  v4u g0;
  g0[0] = 1u;                                        // count=1, gather off
  g0[1] = lds_addr;                                  // LDS byte address
  g0[2] = (unsigned int)ga;                          // global_addr[31:0]
  g0[3] = ((unsigned int)(ga >> 32) & 0x01FFFFFFu) | (2u << 30); // addr[56:32] | type=2
  v8i g1;
  g1[0] = (int)(1u << 16);                           // data_size=1 (2 bytes), mask=0
  g1[1] = (int)((tensor_d0 & 0xFFFFu) << 16);        // tensor_dim0[15:0] @ bits 63:48
  g1[2] = (int)((tensor_d0 >> 16) | ((tensor_d1 & 0xFFFFu) << 16));
  g1[3] = (int)((tensor_d1 >> 16) | ((tile_d0 & 0xFFFFu) << 16)); // tile_dim0 @127:112
  g1[4] = (int)(tile_d1 & 0xFFFFu);                  // tile_dim1; tile_dim2=0
  g1[5] = (int)stride_elems;                         // tensor_dim0_stride[31:0]
  g1[6] = 0;
  g1[7] = 0;
  v4i z4; z4[0]=0; z4[1]=0; z4[2]=0; z4[3]=0;
  v8i z8; for (int i = 0; i < 8; i++) z8[i] = 0;
  __builtin_amdgcn_tensor_load_to_lds(g0, g1, z4, z4, z8, 0);
}

// ---------------------------------------------------------------------------
// Prep / packing kernels (run every call; deterministic)
// ---------------------------------------------------------------------------
__global__ void k_pack_adj(const float* __restrict__ adj, bf16* __restrict__ out){
  int i = blockIdx.x*blockDim.x + threadIdx.x;
  if (i >= 64*64) return;
  int r = i >> 6, c = i & 63;
  out[i] = (bf16)((r < NN && c < NN) ? adj[r*NN + c] : 0.0f);
}

// W_g2 (64,64) -> frag-major 64x64 B (value (k,n) = Wg2[n*64+k])
__global__ void k_pack_w2f(const float* __restrict__ Wg2, bf16* __restrict__ out){
  int i = blockIdx.x*blockDim.x + threadIdx.x;
  if (i >= 4096) return;
  int h = i & 15, lane = (i >> 4) & 31, kc = (i >> 9) & 1, sub = i >> 10;
  int n = sub*16 + (lane & 15);
  int k = kc*32 + ((lane >> 4) << 4) + h;
  out[i] = (bf16)Wg2[n*64 + k];
}

// Wih (2,512,K) -> frag-major B, Npad=1024: blob layout
// [nblk][kc][sub(8)][lane(32)][h(16)], value (k, n=d*512+g) = Wih[(d*512+g)*K + k]
__global__ void k_pack_wih_fm(const float* __restrict__ Wih, bf16* __restrict__ out, int K){
  int i = blockIdx.x*blockDim.x + threadIdx.x;
  if (i >= (K << 10)) return;
  int h = i & 15, lane = (i >> 4) & 31, sub = (i >> 9) & 7;
  int rest = i >> 12;
  int kchunks = K >> 5;
  int kc = rest % kchunks;
  int nblk = rest / kchunks;
  int n = nblk*128 + sub*16 + (lane & 15);
  int k = kc*32 + ((lane >> 4) << 4) + h;
  int d = n >> 9, g = n & 511;
  out[i] = (bf16)Wih[((size_t)(d*512 + g))*K + k];
}

// Wa1 (64,256) -> frag-major B, Npad=128, K=256; zero-pad cols >= 64
__global__ void k_pack_wa1_fm(const float* __restrict__ Wa1, bf16* __restrict__ out){
  int i = blockIdx.x*blockDim.x + threadIdx.x;
  if (i >= 32768) return;
  int h = i & 15, lane = (i >> 4) & 31, sub = (i >> 9) & 7;
  int kc = i >> 12;                     // nblk = 0, 8 k-chunks
  int n = sub*16 + (lane & 15);
  int k = kc*32 + ((lane >> 4) << 4) + h;
  out[i] = (bf16)((n < 64) ? Wa1[n*256 + k] : 0.0f);
}

// Whh (2, 512, 128) -> WMMA-B-fragment-major: [d][ntile][kiter][lane][h]
__global__ void k_pack_whhF(const float* __restrict__ Whh, bf16* __restrict__ out){
  int i = blockIdx.x*blockDim.x + threadIdx.x;
  if (i >= 2*65536) return;
  int d = i >> 16, r = i & 65535;
  int h = r & 15, lane = (r >> 4) & 31, kit = (r >> 9) & 3, nt = r >> 11;
  int g = nt*16 + (lane & 15);
  int k = kit*32 + h + ((lane & 16) ? 16 : 0);
  out[i] = (bf16)Whh[((size_t)(d*512 + g))*LH + k];
}

__global__ void k_bias2(const float* __restrict__ a, const float* __restrict__ b,
                        float* __restrict__ out, int n){
  int i = blockIdx.x*blockDim.x + threadIdx.x;
  if (i < n) out[i] = a[i] + b[i];
}

// ---------------------------------------------------------------------------
// GCN: per sample (b,t): h1 = gelu(adj@(X@W1^T)+b1); h2 = gelu((adj@h1)@W2^T+b2)
// All 64x64x64 matmuls via WMMA bf16; B operands kept frag-major in LDS.
// ---------------------------------------------------------------------------
__global__ __launch_bounds__(128) void k_gcn(const float* __restrict__ x,
                      const bf16* __restrict__ adjb_g, const bf16* __restrict__ w2f_g,
                      const float* __restrict__ Wg1, const float* __restrict__ b1,
                      const float* __restrict__ b2, bf16* __restrict__ seq)
{
  __shared__ __align__(32) bf16 adjs[64*64];  // row-major (A operand)
  __shared__ __align__(32) bf16 w2f [4096];   // frag-major (B operand)
  __shared__ __align__(32) bf16 Yf  [4096];   // frag-major
  __shared__ __align__(32) bf16 Uf  [4096];   // frag-major
  __shared__ __align__(32) bf16 Tm  [64*64];  // row-major (A operand for stage 3)
  __shared__ float Xs[NN*FIN];
  const int tid = threadIdx.x, lane = tid & 31, wid = tid >> 5;
  const size_t s = blockIdx.x;

  { const uint4* a4 = (const uint4*)adjb_g; uint4* d4 = (uint4*)adjs;
    for (int i = tid; i < 512; i += 128) d4[i] = a4[i];
    const uint4* b4 = (const uint4*)w2f_g; uint4* e4 = (uint4*)w2f;
    for (int i = tid; i < 512; i += 128) e4[i] = b4[i]; }
  for (int i = tid; i < NN*FIN; i += 128) Xs[i] = x[s*(NN*FIN) + i];
  __syncthreads();

  // Y = X @ W_g1^T  (61x3 @ 3x64), padded to 64 rows, stored frag-major
  for (int i = tid; i < 4096; i += 128){
    int m = i >> 6, g = i & 63;
    float v = 0.0f;
    if (m < NN){ for (int f = 0; f < FIN; f++) v += Xs[m*FIN + f]*Wg1[g*FIN + f]; }
    Yf[fidx64(m, g)] = (bf16)v;
  }
  __syncthreads();

  // A row-major, B frag-major
  auto mm = [&](const bf16* As, const bf16* Bf, int nt, v8f& acc){
    for (int v = 0; v < 8; v++) acc[v] = 0.0f;
    for (int kc = 0; kc < 2; kc++){
      v16bf af;
      int row = wid*16 + (lane & 15);
      int kb  = kc*32 + ((lane & 16) ? 16 : 0);
      for (int h = 0; h < 8; h++) af[h]   = As[row*64 + kb + h];
      for (int h = 0; h < 8; h++) af[8+h] = As[row*64 + kb + 8 + h];
      const v16bf* bp = (const v16bf*)(Bf + ((nt*2 + kc)*32 + lane)*16);
      v16bf bfr = *bp;
      acc = __builtin_amdgcn_wmma_f32_16x16x32_bf16(false, af, false, bfr, (short)0, acc, false, false);
    }
  };

  // Stage 1: U = gelu(adj @ Y + b1)  -> frag-major Uf
  for (int nt = 0; nt < 4; nt++){
    v8f acc; mm(adjs, Yf, nt, acc);
    int col = nt*16 + (lane & 15);
    int rb  = wid*16 + ((lane & 16) ? 8 : 0);
    for (int v = 0; v < 8; v++) Uf[fidx64(rb + v, col)] = (bf16)geluf(acc[v] + b1[col]);
  }
  __syncthreads();
  // Stage 2: T = adj @ U  -> row-major Tm (A operand next)
  for (int nt = 0; nt < 4; nt++){
    v8f acc; mm(adjs, Uf, nt, acc);
    int col = nt*16 + (lane & 15);
    int rb  = wid*16 + ((lane & 16) ? 8 : 0);
    for (int v = 0; v < 8; v++) Tm[(rb + v)*64 + col] = (bf16)acc[v];
  }
  __syncthreads();
  // Stage 3: seq = gelu(T @ W2^T + b2), store rows < 61
  for (int nt = 0; nt < 4; nt++){
    v8f acc; mm(Tm, w2f, nt, acc);
    int col = nt*16 + (lane & 15);
    int rb  = wid*16 + ((lane & 16) ? 8 : 0);
    for (int v = 0; v < 8; v++){
      int row = rb + v;
      if (row < NN) seq[s*KSEQ + row*GH + col] = (bf16)geluf(acc[v] + b2[col]);
    }
  }
}

// ---------------------------------------------------------------------------
// Generic bf16 GEMM: C(f32) = act(A(MxK) @ Bfm + bias).
// A: row-major, TDM-staged.  B: frag-major blobs (8KB per (nblk,kchunk)),
// TDM-staged as contiguous 1-row tiles. Double-buffered via TENSORcnt.
// Block: 256 threads (8 waves, 2x4), BM=128 BN=128 BK=32.
// ---------------------------------------------------------------------------
#define BM 128
#define BN 128
#define BK 32
__global__ __launch_bounds__(256) void k_gemm(const bf16* __restrict__ A, const bf16* __restrict__ Bfm,
                       const float* __restrict__ bias, float* __restrict__ C,
                       int M, int Nld, int K, int act)
{
  __shared__ __align__(32) bf16 sA[2][BM*BK];
  __shared__ __align__(32) bf16 sB[2][4096];
  const int tid = threadIdx.x, lane = tid & 31, wid = tid >> 5;
  const int mb = blockIdx.x*BM, nb = blockIdx.y*BN;
  const int wr = wid >> 2, wc = wid & 3;
  const int kiters = K >> 5;
  const bf16* Bblk = Bfm + (size_t)blockIdx.y * kiters * 4096;
  v8f acc[4][2];
  for (int i = 0; i < 4; i++) for (int j = 0; j < 2; j++) for (int v = 0; v < 8; v++) acc[i][j][v] = 0.0f;

  if (wid == 0){
    tdm_load_2d((unsigned int)(uintptr_t)&sA[0][0], A + (size_t)mb*K,
                BK, BM, (unsigned int)K, (unsigned int)(M - mb), (unsigned int)K);
    tdm_load_2d((unsigned int)(uintptr_t)&sB[0][0], Bblk, 4096, 1, 4096, 1, 4096);
  }
  for (int i = 0; i < kiters; i++){
    const int buf = i & 1, nxt = buf ^ 1;
    if ((i + 1) < kiters){
      if (wid == 0){
        tdm_load_2d((unsigned int)(uintptr_t)&sA[nxt][0], A + (size_t)mb*K + (i+1)*BK,
                    BK, BM, (unsigned int)(K - (i+1)*BK), (unsigned int)(M - mb), (unsigned int)K);
        tdm_load_2d((unsigned int)(uintptr_t)&sB[nxt][0], Bblk + (size_t)(i+1)*4096, 4096, 1, 4096, 1, 4096);
      }
      __builtin_amdgcn_s_wait_tensorcnt(2);   // wait for chunk i; chunk i+1 stays in flight
    } else {
      __builtin_amdgcn_s_wait_tensorcnt(0);   // last chunk: drain
    }
    __syncthreads();

    v16bf bfr[2];
    for (int nt = 0; nt < 2; nt++)
      bfr[nt] = *(const v16bf*)(&sB[buf][((wc*2 + nt)*32 + lane)*16]);
    for (int mt = 0; mt < 4; mt++){
      v16bf af;
      int row = wr*64 + mt*16 + (lane & 15);
      int kb  = (lane & 16) ? 16 : 0;
      for (int h = 0; h < 8; h++) af[h]   = sA[buf][row*BK + kb + h];
      for (int h = 0; h < 8; h++) af[8+h] = sA[buf][row*BK + kb + 8 + h];
      acc[mt][0] = __builtin_amdgcn_wmma_f32_16x16x32_bf16(false, af, false, bfr[0], (short)0, acc[mt][0], false, false);
      acc[mt][1] = __builtin_amdgcn_wmma_f32_16x16x32_bf16(false, af, false, bfr[1], (short)0, acc[mt][1], false, false);
    }
    __syncthreads();
  }

  for (int mt = 0; mt < 4; mt++) for (int nt = 0; nt < 2; nt++){
    int col = nb + wc*32 + nt*16 + (lane & 15);
    if (col < Nld){
      float bv = bias ? bias[col] : 0.0f;
      int rbase = mb + wr*64 + mt*16 + ((lane & 16) ? 8 : 0);
      for (int v = 0; v < 8; v++){
        float val = acc[mt][nt][v] + bv;
        if (act == 1) val = tanhf(val);
        C[(size_t)(rbase + v)*Nld + col] = val;
      }
    }
  }
}

// ---------------------------------------------------------------------------
// BiLSTM recurrence: one persistent block per direction (grid=2), 32 waves.
// Per step: G = h @ Whh^T via WMMA (B frags frag-major in global, 32B loads),
// + pre[t], cell math in VALU, c in registers, h in LDS bf16.
// ---------------------------------------------------------------------------
__global__ __launch_bounds__(1024) void k_lstm(const float* __restrict__ pre,
                        const bf16* __restrict__ whhF, bf16* __restrict__ out)
{
  const int d = blockIdx.x;
  const int tid = threadIdx.x, lane = tid & 31, w = tid >> 5;
  __shared__ __align__(32) bf16 hb[64*LH];     // 16 KB
  __shared__ __align__(32) bf16 Gb[32*512];    // 32 KB (half batch of gates)
  for (int i = tid; i < 64*LH; i += 1024) hb[i] = (bf16)0.0f;
  float creg[2][4];
  for (int a = 0; a < 2; a++) for (int q = 0; q < 4; q++) creg[a][q] = 0.0f;
  __syncthreads();
  const bf16* wf = whhF + (size_t)d*65536;

  for (int step = 0; step < T_; step++){
    int tt = (d == 0) ? step : (T_ - 1 - step);
    for (int mh = 0; mh < 2; mh++){
      for (int mt = 0; mt < 2; mt++){
        v8f acc; for (int v = 0; v < 8; v++) acc[v] = 0.0f;
        for (int kit = 0; kit < 4; kit++){
          v16bf af;
          int row = mh*32 + mt*16 + (lane & 15);
          int kb  = kit*32 + ((lane & 16) ? 16 : 0);
          for (int h = 0; h < 8; h++) af[h]   = hb[row*LH + kb + h];
          for (int h = 0; h < 8; h++) af[8+h] = hb[row*LH + kb + 8 + h];
          const v16bf* bp = (const v16bf*)(wf + (((size_t)w*4 + kit)*32 + lane)*16);
          v16bf bfr = *bp;
          acc = __builtin_amdgcn_wmma_f32_16x16x32_bf16(false, af, false, bfr, (short)0, acc, false, false);
        }
        int g  = w*16 + (lane & 15);
        int rb = mt*16 + ((lane & 16) ? 8 : 0);
        for (int v = 0; v < 8; v++){
          int r = rb + v, b = mh*32 + r;
          float val = acc[v] + pre[((size_t)(b*T_ + tt))*NG2 + d*512 + g];
          Gb[r*512 + g] = (bf16)val;
        }
      }
      __syncthreads();
      for (int q = 0; q < 4; q++){
        int idx = q*1024 + tid;
        int r = idx >> 7, j = idx & 127;
        float gi = (float)Gb[r*512 + j];
        float gf = (float)Gb[r*512 + 128 + j];
        float gc = (float)Gb[r*512 + 256 + j];
        float go = (float)Gb[r*512 + 384 + j];
        float c  = sigf(gf)*creg[mh][q] + sigf(gi)*tanhf(gc);
        creg[mh][q] = c;
        float h  = sigf(go)*tanhf(c);
        int b = mh*32 + r;
        hb[b*LH + j] = (bf16)h;
        out[((size_t)(b*T_ + tt))*256 + d*LH + j] = (bf16)h;
      }
      __syncthreads();
    }
  }
}

// ---------------------------------------------------------------------------
// Attention + MLP head: one block per batch element.
// ---------------------------------------------------------------------------
__global__ __launch_bounds__(256) void k_attn(const float* __restrict__ S1, const bf16* __restrict__ l1o,
                       const float* __restrict__ Wa2, const float* __restrict__ ba2,
                       const float* __restrict__ Wc1, const float* __restrict__ bc1,
                       const float* __restrict__ Wc2, const float* __restrict__ bc2,
                       float* __restrict__ outp)
{
  __shared__ float sw[T_];
  __shared__ float red[T_];
  const int t = threadIdx.x, b = blockIdx.x;
  float sc = ba2[0];
  const float* s1r = S1 + ((size_t)(b*T_) + t)*64;
  for (int k = 0; k < 64; k++) sc += s1r[k]*Wa2[k];
  sw[t] = sc; red[t] = sc; __syncthreads();
  for (int st = 128; st > 0; st >>= 1){ if (t < st) red[t] = fmaxf(red[t], red[t+st]); __syncthreads(); }
  float mx = red[0]; __syncthreads();
  float e = expf(sc - mx); sw[t] = e; red[t] = e; __syncthreads();
  for (int st = 128; st > 0; st >>= 1){ if (t < st) red[t] += red[t+st]; __syncthreads(); }
  float inv = 1.0f/red[0]; __syncthreads();
  float acc = 0.0f;
  for (int tt = 0; tt < T_; tt++) acc += sw[tt]*(float)l1o[((size_t)(b*T_ + tt))*256 + t];
  acc *= inv;
  red[t] = acc; __syncthreads();
  if (t < LH){
    float hv = bc1[t];
    for (int k = 0; k < 256; k++) hv += red[k]*Wc1[t*256 + k];
    sw[t] = fmaxf(hv, 0.0f);
  }
  __syncthreads();
  if (t < NCLS){
    float o = bc2[t];
    for (int k = 0; k < LH; k++) o += sw[k]*Wc2[t*LH + k];
    outp[b*NCLS + t] = o;
  }
}

// ---------------------------------------------------------------------------
extern "C" void kernel_launch(void* const* d_in, const int* in_sizes, int n_in,
                              void* d_out, int out_size, void* d_ws, size_t ws_size,
                              hipStream_t stream)
{
  (void)in_sizes; (void)n_in; (void)out_size; (void)ws_size;
  const float* x    = (const float*)d_in[0];
  const float* adj  = (const float*)d_in[1];
  const float* Wg1  = (const float*)d_in[2];
  const float* bg1  = (const float*)d_in[3];
  const float* Wg2  = (const float*)d_in[4];
  const float* bg2  = (const float*)d_in[5];
  const float* Wih0 = (const float*)d_in[6];
  const float* Whh0 = (const float*)d_in[7];
  const float* bih0 = (const float*)d_in[8];
  const float* bhh0 = (const float*)d_in[9];
  const float* Wih1 = (const float*)d_in[10];
  const float* Whh1 = (const float*)d_in[11];
  const float* bih1 = (const float*)d_in[12];
  const float* bhh1 = (const float*)d_in[13];
  const float* Wa1  = (const float*)d_in[14];
  const float* ba1  = (const float*)d_in[15];
  const float* Wa2  = (const float*)d_in[16];
  const float* ba2  = (const float*)d_in[17];
  const float* Wc1  = (const float*)d_in[18];
  const float* bc1  = (const float*)d_in[19];
  const float* Wc2  = (const float*)d_in[20];
  const float* bc2  = (const float*)d_in[21];
  float* outp = (float*)d_out;

  char* w = (char*)d_ws;
  size_t o = 0;
  auto al = [&](size_t bytes){ size_t r = o; o = (o + bytes + 255) & ~(size_t)255; return r; };
  bf16*  adjb   = (bf16*)(w + al((size_t)64*64*2));
  bf16*  w2f    = (bf16*)(w + al((size_t)4096*2));
  bf16*  wa1f   = (bf16*)(w + al((size_t)32768*2));
  bf16*  w0F    = (bf16*)(w + al((size_t)KSEQ*1024*2));
  bf16*  w1F    = (bf16*)(w + al((size_t)256*1024*2));
  bf16*  whhF0  = (bf16*)(w + al((size_t)2*65536*2));
  bf16*  whhF1  = (bf16*)(w + al((size_t)2*65536*2));
  float* bias01 = (float*)(w + al((size_t)1024*4));
  float* bias11 = (float*)(w + al((size_t)1024*4));
  bf16*  seqb   = (bf16*)(w + al((size_t)S_*KSEQ*2));
  float* pre0   = (float*)(w + al((size_t)S_*1024*4));
  bf16*  l0o    = (bf16*)(w + al((size_t)S_*256*2));
  float* pre1   = (float*)(w + al((size_t)S_*1024*4));
  bf16*  l1o    = (bf16*)(w + al((size_t)S_*256*2));
  float* S1f    = (float*)(w + al((size_t)S_*64*4));

  // ---- prep / packing ----
  k_pack_adj   <<<16, 256, 0, stream>>>(adj, adjb);
  k_pack_w2f   <<<16, 256, 0, stream>>>(Wg2, w2f);
  k_pack_wa1_fm<<<128, 256, 0, stream>>>(Wa1, wa1f);
  k_pack_wih_fm<<<(KSEQ*1024)/256, 256, 0, stream>>>(Wih0, w0F, KSEQ);
  k_pack_wih_fm<<<(256*1024)/256, 256, 0, stream>>>(Wih1, w1F, 256);
  k_pack_whhF  <<<512, 256, 0, stream>>>(Whh0, whhF0);
  k_pack_whhF  <<<512, 256, 0, stream>>>(Whh1, whhF1);
  k_bias2      <<<4, 256, 0, stream>>>(bih0, bhh0, bias01, 1024);
  k_bias2      <<<4, 256, 0, stream>>>(bih1, bhh1, bias11, 1024);

  // ---- GCN front-end (WMMA per sample) ----
  k_gcn<<<S_, 128, 0, stream>>>(x, adjb, w2f, Wg1, bg1, bg2, seqb);

  // ---- BiLSTM layer 0 ----
  k_gemm<<<dim3(S_/BM, NG2/BN), 256, 0, stream>>>(seqb, w0F, bias01, pre0, S_, NG2, KSEQ, 0);
  k_lstm<<<2, 1024, 0, stream>>>(pre0, whhF0, l0o);

  // ---- BiLSTM layer 1 ----
  k_gemm<<<dim3(S_/BM, NG2/BN), 256, 0, stream>>>(l0o, w1F, bias11, pre1, S_, NG2, 256, 0);
  k_lstm<<<2, 1024, 0, stream>>>(pre1, whhF1, l1o);

  // ---- attention scores (tanh GEMM, N=64: packed zero-pad + store guard) ----
  k_gemm<<<dim3(S_/BM, 1), 256, 0, stream>>>(l1o, wa1f, ba1, S1f, S_, 64, 256, 1);

  // ---- softmax attention + MLP head ----
  k_attn<<<B_, 256, 0, stream>>>(S1f, l1o, Wa2, ba2, Wc1, bc1, Wc2, bc2, outp);
}
